// QuantizedSqueezeExcitation_75488345195240
// MI455X (gfx1250) — matrix operand order repaired and checked by reading, not compiled, not run on hardware
//
#include <hip/hip_runtime.h>
#include <hip/hip_bf16.h>

typedef __attribute__((ext_vector_type(16))) _Float16 v16h;
typedef __attribute__((ext_vector_type(8)))  float    v8f;
typedef __attribute__((ext_vector_type(4)))  float    f32x4;

#define C_IN   256
#define C_SQ   64
#define BATCH  32
#define HW     3136               // 56*56
#define HW4    (HW / 4)           // 784 float4 per plane
#define PLANES (BATCH * C_IN)     // 8192

// ---------------------------------------------------------------------------
// Kernel 1: global average pool. One block per (b,c) plane. Regular-temporal
// 128-bit loads intentionally fill L2 with x so the excite pass re-reads from
// L2 instead of HBM (x = 102.8 MB < 192 MB L2).
// ---------------------------------------------------------------------------
__global__ __launch_bounds__(256) void se_pool(const float* __restrict__ x,
                                               float* __restrict__ pooled) {
    __shared__ float red[256];
    const int plane = blockIdx.x;
    const f32x4* xp = (const f32x4*)(x + (size_t)plane * HW);
    float sum = 0.0f;
    for (int i = threadIdx.x; i < HW4; i += 256) {
        f32x4 v = xp[i];
        sum += v[0] + v[1] + v[2] + v[3];
    }
    red[threadIdx.x] = sum;
    __syncthreads();
    #pragma unroll
    for (int s = 128; s > 0; s >>= 1) {
        if ((int)threadIdx.x < s) red[threadIdx.x] += red[threadIdx.x + s];
        __syncthreads();
    }
    if (threadIdx.x == 0) pooled[plane] = red[0] * (1.0f / (float)HW);
}

// ---------------------------------------------------------------------------
// Kernel 2: fc1 + fc2 + hardsigmoid on the WMMA pipeline.
// Single block, 8 wave32s. v_wmma_f32_16x16x32_f16, f32 accumulate.
//   GEMM1: s1[32x64]  = pooled[32x256] @ w1^T + b1     (8 tiles, 1 per wave)
//   GEMM2: sc[32x256] = s1[32x64] @ w2^T + b2, hsig    (32 tiles, 4 per wave)
// Fragment packing uses contiguous k (k = 16*half + s) for A and B, not the ISA table order.
// ---------------------------------------------------------------------------
__global__ __launch_bounds__(256) void se_fc(const float* __restrict__ pooled,
                                             const float* __restrict__ w1,
                                             const float* __restrict__ b1,
                                             const float* __restrict__ w2,
                                             const float* __restrict__ b2,
                                             float* __restrict__ scales) {
    __shared__ _Float16 s1[BATCH][C_SQ];   // fc1 output handed to GEMM2 (4 KB)

    const int lane = threadIdx.x & 31;
    const int wave = threadIdx.x >> 5;     // 0..7
    const int hi   = lane >> 4;            // which 16-lane half
    const int l16  = lane & 15;

    // ---- GEMM1: M=32, N=64, K=256 -> 2x4 tiles, one per wave --------------
    {
        const int mt = wave >> 2;          // 0..1
        const int nt = wave & 3;           // 0..3
        v8f acc = {};
        #pragma unroll
        for (int ks = 0; ks < 8; ++ks) {
            const int k0 = ks * 32;
            v16h a, b;
            #pragma unroll
            for (int j = 0; j < 8; ++j) {
                // A packed in contiguous k order: lanes<16 K=0..15, lanes>=16 K=16..31
                const int ka = k0 + hi * 16 + 2 * j;
                const float* arow = pooled + (mt * 16 + l16) * C_IN;
                a[2 * j]     = (_Float16)arow[ka];
                a[2 * j + 1] = (_Float16)arow[ka + 1];
                // B packed the same way: lanes<16 K=0..15, lanes>=16 K=16..31
                // B[k][n] = w1[n][k]  (einsum 'bc,sc->bs')
                const int kb = k0 + hi * 16 + 2 * j;
                const float* brow = w1 + (nt * 16 + l16) * C_IN;
                b[2 * j]     = (_Float16)brow[kb];
                b[2 * j + 1] = (_Float16)brow[kb + 1];
            }
            acc = __builtin_amdgcn_wmma_f32_16x16x32_f16(
                false, a, false, b, (short)0, acc, false, false);
        }
        // C/D layout: VGPR r -> M = r (lanes 0-15) / r+8 (lanes 16-31), N = l16
        const int ng = nt * 16 + l16;
        const float bias = b1[ng];
        #pragma unroll
        for (int r = 0; r < 8; ++r) {
            const int m = mt * 16 + r + 8 * hi;
            s1[m][ng] = (_Float16)(acc[r] + bias);
        }
    }
    __syncthreads();

    // ---- GEMM2: M=32, N=256, K=64 -> 2x16 tiles, 4 per wave ---------------
    #pragma unroll
    for (int t = 0; t < 4; ++t) {
        const int tile = wave * 4 + t;     // 0..31
        const int mt = tile >> 4;          // 0..1
        const int nt = tile & 15;          // 0..15
        v8f acc = {};
        #pragma unroll
        for (int ks = 0; ks < 2; ++ks) {
            const int k0 = ks * 32;
            v16h a, b;
            #pragma unroll
            for (int j = 0; j < 8; ++j) {
                const int ka = k0 + hi * 16 + 2 * j;
                a[2 * j]     = s1[mt * 16 + l16][ka];
                a[2 * j + 1] = s1[mt * 16 + l16][ka + 1];
                // B[k][n] = w2[n][k]  (einsum 'bs,cs->bc')
                const int kb = k0 + hi * 16 + 2 * j;
                const float* brow = w2 + (nt * 16 + l16) * C_SQ;
                b[2 * j]     = (_Float16)brow[kb];
                b[2 * j + 1] = (_Float16)brow[kb + 1];
            }
            acc = __builtin_amdgcn_wmma_f32_16x16x32_f16(
                false, a, false, b, (short)0, acc, false, false);
        }
        const int cg = nt * 16 + l16;
        const float bias = b2[cg];
        #pragma unroll
        for (int r = 0; r < 8; ++r) {
            const int m = mt * 16 + r + 8 * hi;
            float v = acc[r] + bias;
            v = fminf(fmaxf(v + 3.0f, 0.0f), 6.0f) * (1.0f / 6.0f);  // hardsigmoid
            scales[m * C_IN + cg] = v;
        }
    }
}

// ---------------------------------------------------------------------------
// Kernel 3: out = x * scale[b,c]. 128-bit NT loads (x already resident in L2
// from the pool pass; NT marks lines for early eviction after this last use)
// and NT stores (don't let the 103 MB output evict x mid-pass).
// ---------------------------------------------------------------------------
__global__ __launch_bounds__(256) void se_excite(const float* __restrict__ x,
                                                 const float* __restrict__ scales,
                                                 float* __restrict__ out,
                                                 unsigned total4) {
    const unsigned i = blockIdx.x * 256u + threadIdx.x;
    if (i >= total4) return;
    const unsigned plane = i / (unsigned)HW4;  // HW % 4 == 0 -> no straddle
    const float s = scales[plane];
    f32x4 v = __builtin_nontemporal_load((const f32x4*)x + i);
    v *= s;
    __builtin_nontemporal_store(v, (f32x4*)out + i);
}

extern "C" void kernel_launch(void* const* d_in, const int* in_sizes, int n_in,
                              void* d_out, int out_size, void* d_ws, size_t ws_size,
                              hipStream_t stream) {
    (void)in_sizes; (void)n_in; (void)out_size; (void)ws_size;
    const float* x  = (const float*)d_in[0];
    const float* w1 = (const float*)d_in[1];
    const float* b1 = (const float*)d_in[2];
    const float* w2 = (const float*)d_in[3];
    const float* b2 = (const float*)d_in[4];
    float* out    = (float*)d_out;
    float* pooled = (float*)d_ws;           // [32*256] f32
    float* scales = pooled + PLANES;        // [32*256] f32

    se_pool<<<PLANES, 256, 0, stream>>>(x, pooled);
    se_fc<<<1, 256, 0, stream>>>(pooled, w1, b1, w2, b2, scales);

    const unsigned total4 = (unsigned)PLANES * (unsigned)HW4;  // 6,422,528
    se_excite<<<(total4 + 255u) / 256u, 256, 0, stream>>>(x, scales, out, total4);
}
